// HealthGNN_70549132804339
// MI455X (gfx1250) — matrix operand order repaired and checked by reading, not compiled, not run on hardware
//
#include <hip/hip_runtime.h>
#include <hip/hip_bf16.h>
#include <math.h>

typedef __attribute__((ext_vector_type(16))) __bf16 v16bf;
typedef __attribute__((ext_vector_type(8)))  float  v8f;

#define NNODES 100000
#define NEDGES 3200000
#define NGRAPH 256
#define HDIM   64
#define EPSBN  1e-5f

// ---------------------------------------------------------------- degree prep
__global__ void deg_init_kernel(float* __restrict__ deg, int n) {
    int i = blockIdx.x * blockDim.x + threadIdx.x;
    if (i < n) deg[i] = 1.0f;   // self-loop contributes 1
}

__global__ void deg_accum_kernel(const int* __restrict__ dst, float* __restrict__ deg, int e) {
    int i = blockIdx.x * blockDim.x + threadIdx.x;
    if (i < e) unsafeAtomicAdd(&deg[dst[i]], 1.0f);
}

__global__ void dinv_kernel(float* __restrict__ deg, int n) {
    int i = blockIdx.x * blockDim.x + threadIdx.x;
    if (i < n) deg[i] = rsqrtf(deg[i]);   // in-place: deg -> dinv
}

// ---------------------------------------------------------------- weight pack
// Pack W (K x hout, f32, row-major) into bf16 B-fragments:
//   out[((chunk*(hout/16) + wave)*32 + lane)*16 + j]
//   lane: hi = lane>>4, n = lane&15, col = wave*16+n, k = chunk*32 + (j<8 ? 8*hi+j : 16+8*hi+(j-8))
__global__ void pack_w_kernel(const float* __restrict__ W, __bf16* __restrict__ out,
                              int K, int hout, int nchunks) {
    int idx = blockIdx.x * blockDim.x + threadIdx.x;
    int total = nchunks * (hout / 16) * 32 * 16;
    if (idx >= total) return;
    int j    = idx & 15;
    int lane = (idx >> 4) & 31;
    int wv   = (idx >> 9) % (hout / 16);
    int ch   = idx / (512 * (hout / 16));
    int hi   = lane >> 4;
    int col  = wv * 16 + (lane & 15);
    int k    = ch * 32 + ((j < 8) ? (8 * hi + j) : (16 + 8 * hi + (j - 8)));
    float v  = (k < K) ? W[k * hout + col] : 0.0f;
    out[idx] = (__bf16)v;
}

// ---------------------------------------------------------------- WMMA GEMM
// T = A(nrows x K, f32) @ W(K x HOUT) via bf16 WMMA, f32 accumulate.
// B comes pre-packed in fragment order. nrows MUST be a multiple of 16.
// MODE 0: t_out = T ; agg_out = T * dinv[row]^2 + bias[col]   (GCN seed)
// MODE 1: t_out = relu(T + bias[col])                          (FC layer)
// Launch: grid = nrows/16, blockDim = 32 * (HOUT/16).
template <int K, int HOUT, int MODE>
__global__ void wmma_gemm_kernel(const float* __restrict__ A,
                                 const v16bf* __restrict__ Bpk,
                                 const float* __restrict__ bias,
                                 const float* __restrict__ dinv,
                                 float* __restrict__ t_out,
                                 float* __restrict__ agg_out) {
    constexpr int NCH = (K + 31) / 32;
    constexpr int NW  = HOUT / 16;
    const int lane = threadIdx.x & 31;
    const int wave = threadIdx.x >> 5;
    const int tile = blockIdx.x;
    const int mloc = lane & 15;
    const int hi   = lane >> 4;               // lane half: 0 or 1
    const int row  = tile * 16 + mloc;        // A row for this lane
    const int col  = wave * 16 + mloc;        // B/C/D column for this lane

    v8f acc = {};
#pragma unroll
    for (int c = 0; c < NCH; ++c) {
        const int k0 = c * 32;
        v16bf a;
        if constexpr (K % 32 == 0) {
            // Contiguous, unconditional float4 loads (A row base is 128B aligned).
            const float4* ar = (const float4*)(A + (size_t)row * K + k0);
            float av[16];
            *(float4*)(av + 0)  = ar[2 * hi];          // k = k0+8*hi   .. +3
            *(float4*)(av + 4)  = ar[2 * hi + 1];      // k = k0+8*hi+4 .. +7
            *(float4*)(av + 8)  = ar[4 + 2 * hi];      // k = k0+16+8*hi    ..
            *(float4*)(av + 12) = ar[4 + 2 * hi + 1];  // k = k0+16+8*hi+4  ..
#pragma unroll
            for (int j = 0; j < 16; ++j) a[j] = (__bf16)av[j];
        } else {
            // Clamped unconditional addresses + value masking (no exec branches).
            const int rowbase = row * K;
#pragma unroll
            for (int j = 0; j < 16; ++j) {
                int k  = k0 + ((j < 8) ? (8 * hi + j) : (16 + 8 * hi + (j - 8)));
                int kc = (k < K) ? k : 0;
                float v = A[rowbase + kc];
                a[j] = (__bf16)((k < K) ? v : 0.0f);
            }
        }
        v16bf b = Bpk[(c * NW + wave) * 32 + lane];   // one 32B fragment load
        acc = __builtin_amdgcn_wmma_f32_16x16x32_bf16(false, a, false, b,
                                                      (short)0, acc, false, false);
    }

    // D layout: VGPR r -> M = r + 8*hi, N = lane&15
    const float bcol = bias[col];
#pragma unroll
    for (int r = 0; r < 8; ++r) {
        int orow = tile * 16 + r + 8 * hi;
        size_t o = (size_t)orow * HOUT + col;
        float v  = acc[r];
        if constexpr (MODE == 0) {
            t_out[o] = v;
            float di = dinv[orow];
            agg_out[o] = v * di * di + bcol;
        } else {
            float t = v + bcol;
            t_out[o] = t > 0.0f ? t : 0.0f;
        }
    }
}

// ---------------------------------------------------------------- edge scatter
// 16 threads per edge, each handles 4 consecutive channels (float4 gather,
// 4 hardware f32 atomics). Hot data (t: 25.6MB, edges: 25.6MB) is L2-resident.
__global__ void scatter_kernel(const int* __restrict__ src, const int* __restrict__ dst,
                               const float* __restrict__ dinv, const float* __restrict__ t,
                               float* __restrict__ agg, int e) {
    long long gid = (long long)blockIdx.x * blockDim.x + threadIdx.x;
    if (gid >= (long long)e * 16) return;
    int edge = (int)(gid >> 4);
    int c    = (int)(gid & 15) * 4;
    int s = src[edge];
    int d = dst[edge];
    float en = dinv[s] * dinv[d];
    const float4 v = *(const float4*)(t + (long long)s * HDIM + c);
    float* base = agg + (long long)d * HDIM + c;
    unsafeAtomicAdd(base + 0, v.x * en);
    unsafeAtomicAdd(base + 1, v.y * en);
    unsafeAtomicAdd(base + 2, v.z * en);
    unsafeAtomicAdd(base + 3, v.w * en);
}

// ---------------------------------------------------------------- BN + ReLU (float4)
__device__ __forceinline__ float4 bn4(float4 v, float4 G, float4 B, float4 M, float4 V) {
    float4 o;
    o.x = fmaxf((v.x - M.x) * rsqrtf(V.x + EPSBN) * G.x + B.x, 0.0f);
    o.y = fmaxf((v.y - M.y) * rsqrtf(V.y + EPSBN) * G.y + B.y, 0.0f);
    o.z = fmaxf((v.z - M.z) * rsqrtf(V.z + EPSBN) * G.z + B.z, 0.0f);
    o.w = fmaxf((v.w - M.w) * rsqrtf(V.w + EPSBN) * G.w + B.w, 0.0f);
    return o;
}

__global__ void bnrelu4_kernel(const float* __restrict__ agg,
                               const float* __restrict__ g,  const float* __restrict__ be,
                               const float* __restrict__ rm, const float* __restrict__ rv,
                               float* __restrict__ h, int n) {
    int gid = blockIdx.x * blockDim.x + threadIdx.x;
    if (gid >= n * 16) return;
    int i = gid >> 4;
    int c = (gid & 15) * 4;
    float4 v = *(const float4*)(agg + (size_t)i * HDIM + c);
    float4 o = bn4(v, *(const float4*)(g + c), *(const float4*)(be + c),
                      *(const float4*)(rm + c), *(const float4*)(rv + c));
    *(float4*)(h + (size_t)i * HDIM + c) = o;
}

// Layer-3 fusion: BN+ReLU -> atomic mean-pool accumulate. h3 never hits memory.
__global__ void bnrelu_pool_kernel(const float* __restrict__ agg,
                                   const float* __restrict__ g,  const float* __restrict__ be,
                                   const float* __restrict__ rm, const float* __restrict__ rv,
                                   const int* __restrict__ batch,
                                   float* __restrict__ pooled, int n) {
    int gid = blockIdx.x * blockDim.x + threadIdx.x;
    if (gid >= n * 16) return;
    int i = gid >> 4;
    int c = (gid & 15) * 4;
    float4 v = *(const float4*)(agg + (size_t)i * HDIM + c);
    float4 o = bn4(v, *(const float4*)(g + c), *(const float4*)(be + c),
                      *(const float4*)(rm + c), *(const float4*)(rv + c));
    float* base = pooled + (size_t)batch[i] * HDIM + c;
    unsafeAtomicAdd(base + 0, o.x);
    unsafeAtomicAdd(base + 1, o.y);
    unsafeAtomicAdd(base + 2, o.z);
    unsafeAtomicAdd(base + 3, o.w);
}

// ---------------------------------------------------------------- pooling helpers
__global__ void zero_kernel(float* __restrict__ p, int n) {
    int i = blockIdx.x * blockDim.x + threadIdx.x;
    if (i < n) p[i] = 0.0f;
}

__global__ void count_kernel(const int* __restrict__ batch, float* __restrict__ counts, int n) {
    int i = blockIdx.x * blockDim.x + threadIdx.x;
    if (i < n) unsafeAtomicAdd(&counts[batch[i]], 1.0f);
}

__global__ void pool_div_kernel(float* __restrict__ pooled, const float* __restrict__ counts) {
    int gid = blockIdx.x * blockDim.x + threadIdx.x;
    if (gid < NGRAPH * HDIM) {
        float cnt = counts[gid >> 6];
        pooled[gid] /= fmaxf(cnt, 1.0f);
    }
}

// ---------------------------------------------------------------- final FC2 + sigmoid
__global__ void fc2_kernel(const float* __restrict__ fc1, const float* __restrict__ Wf2,
                           const float* __restrict__ bf2, float* __restrict__ out) {
    int gid = blockIdx.x * blockDim.x + threadIdx.x;
    if (gid < NGRAPH * 2) {
        int g = gid >> 1;
        int o = gid & 1;
        float s = bf2[o];
#pragma unroll
        for (int j = 0; j < 32; ++j) s += fc1[g * 32 + j] * Wf2[j * 2 + o];
        out[gid] = 1.0f / (1.0f + expf(-s));
    }
}

// ================================================================ launcher
extern "C" void kernel_launch(void* const* d_in, const int* in_sizes, int n_in,
                              void* d_out, int out_size, void* d_ws, size_t ws_size,
                              hipStream_t stream) {
    const float* x   = (const float*)d_in[0];
    const int*   ei  = (const int*)d_in[1];      // [2,E]: src then dst
    const int*   bat = (const int*)d_in[2];
    const float* W[3]  = { (const float*)d_in[3],  (const float*)d_in[9],  (const float*)d_in[15] };
    const float* b[3]  = { (const float*)d_in[4],  (const float*)d_in[10], (const float*)d_in[16] };
    const float* g[3]  = { (const float*)d_in[5],  (const float*)d_in[11], (const float*)d_in[17] };
    const float* be[3] = { (const float*)d_in[6],  (const float*)d_in[12], (const float*)d_in[18] };
    const float* rm[3] = { (const float*)d_in[7],  (const float*)d_in[13], (const float*)d_in[19] };
    const float* rv[3] = { (const float*)d_in[8],  (const float*)d_in[14], (const float*)d_in[20] };
    const float* Wf1 = (const float*)d_in[21];
    const float* bf1 = (const float*)d_in[22];
    const float* Wf2 = (const float*)d_in[23];
    const float* bf2 = (const float*)d_in[24];
    float* out = (float*)d_out;

    const int* src = ei;
    const int* dst = ei + NEDGES;

    float* ws     = (float*)d_ws;
    float* dinv   = ws;                                   // N
    float* t      = dinv + NNODES;                        // N*64
    float* agg    = t + (size_t)NNODES * HDIM;            // N*64
    float* h      = agg + (size_t)NNODES * HDIM;          // N*64
    float* pooled = h + (size_t)NNODES * HDIM;            // G*64
    float* counts = pooled + NGRAPH * HDIM;               // G
    float* fc1o   = counts + NGRAPH;                      // G*32
    // packed bf16 weight fragments (32B-aligned region)
    __bf16* pk1  = (__bf16*)(fc1o + NGRAPH * 32);         // 1 chunk * 4 waves * 512
    __bf16* pk2  = pk1 + 2048;                            // 2 chunks * 4 waves * 512
    __bf16* pk3  = pk2 + 4096;
    __bf16* pkf1 = pk3 + 4096;                            // 2 chunks * 2 waves * 512

    const int TB = 256;

    // pack weights into B-fragment layout (cheap one-shot kernels)
    pack_w_kernel<<<(2048 + TB - 1) / TB, TB, 0, stream>>>(W[0], pk1,  5,  HDIM, 1);
    pack_w_kernel<<<(4096 + TB - 1) / TB, TB, 0, stream>>>(W[1], pk2,  64, HDIM, 2);
    pack_w_kernel<<<(4096 + TB - 1) / TB, TB, 0, stream>>>(W[2], pk3,  64, HDIM, 2);
    pack_w_kernel<<<(2048 + TB - 1) / TB, TB, 0, stream>>>(Wf1,  pkf1, 64, 32,   2);

    // degree / dinv (self-loop included)
    deg_init_kernel<<<(NNODES + TB - 1) / TB, TB, 0, stream>>>(dinv, NNODES);
    deg_accum_kernel<<<(NEDGES + TB - 1) / TB, TB, 0, stream>>>(dst, dinv, NEDGES);
    dinv_kernel<<<(NNODES + TB - 1) / TB, TB, 0, stream>>>(dinv, NNODES);

    // zero pooled + counts early (contiguous); counts needs only the batch array
    zero_kernel<<<(NGRAPH * HDIM + NGRAPH + TB - 1) / TB, TB, 0, stream>>>(
        pooled, NGRAPH * HDIM + NGRAPH);
    count_kernel<<<(NNODES + TB - 1) / TB, TB, 0, stream>>>(bat, counts, NNODES);

    const int sblocks = (int)(((long long)NEDGES * 16 + TB - 1) / TB);
    const int nb16    = (NNODES * 16 + TB - 1) / TB;

    // ---- layer 1 (K=5)
    wmma_gemm_kernel<5, HDIM, 0><<<NNODES / 16, 32 * (HDIM / 16), 0, stream>>>(
        x, (const v16bf*)pk1, b[0], dinv, t, agg);
    scatter_kernel<<<sblocks, TB, 0, stream>>>(src, dst, dinv, t, agg, NEDGES);
    bnrelu4_kernel<<<nb16, TB, 0, stream>>>(agg, g[0], be[0], rm[0], rv[0], h, NNODES);

    // ---- layer 2 (K=64)
    wmma_gemm_kernel<HDIM, HDIM, 0><<<NNODES / 16, 32 * (HDIM / 16), 0, stream>>>(
        h, (const v16bf*)pk2, b[1], dinv, t, agg);
    scatter_kernel<<<sblocks, TB, 0, stream>>>(src, dst, dinv, t, agg, NEDGES);
    bnrelu4_kernel<<<nb16, TB, 0, stream>>>(agg, g[1], be[1], rm[1], rv[1], h, NNODES);

    // ---- layer 3 (K=64): BN+ReLU fused with pooling accumulation
    wmma_gemm_kernel<HDIM, HDIM, 0><<<NNODES / 16, 32 * (HDIM / 16), 0, stream>>>(
        h, (const v16bf*)pk3, b[2], dinv, t, agg);
    scatter_kernel<<<sblocks, TB, 0, stream>>>(src, dst, dinv, t, agg, NEDGES);
    bnrelu_pool_kernel<<<nb16, TB, 0, stream>>>(
        agg, g[2], be[2], rm[2], rv[2], bat, pooled, NNODES);

    // ---- mean pooling divide
    pool_div_kernel<<<(NGRAPH * HDIM + TB - 1) / TB, TB, 0, stream>>>(pooled, counts);

    // ---- FC1 (256x64 @ 64x32, +bias, ReLU) via WMMA
    wmma_gemm_kernel<HDIM, 32, 1><<<NGRAPH / 16, 32 * (32 / 16), 0, stream>>>(
        pooled, (const v16bf*)pkf1, bf1, nullptr, fc1o, nullptr);

    // ---- FC2 + sigmoid
    fc2_kernel<<<(NGRAPH * 2 + TB - 1) / TB, TB, 0, stream>>>(fc1o, Wf2, bf2, out);
}
